// MultiHeadAttentionCombiner_34789235097861
// MI455X (gfx1250) — matrix operand order repaired and checked by reading, not compiled, not run on hardware
//
#include <hip/hip_runtime.h>

#define NB    32
#define CC    256
#define HWPIX 4096
#define CMDD  768
#define HEADS 8
#define DH    64
#define INNER 512

typedef __attribute__((ext_vector_type(16))) __bf16 v16bf;
typedef __attribute__((ext_vector_type(8)))  float  v8f;
typedef __attribute__((ext_vector_type(4)))  unsigned int u32x4;

union FragBF { u32x4 q[2]; v16bf v; };

__device__ __forceinline__ unsigned short f2bf(float x) {
  union { float f; unsigned u; } c; c.f = x;
  unsigned r = c.u + 0x7FFFu + ((c.u >> 16) & 1u);
  return (unsigned short)(r >> 16);
}
__device__ __forceinline__ float bf2f(unsigned short u) {
  union { unsigned u; float f; } c; c.u = ((unsigned)u) << 16;
  return c.f;
}
__device__ __forceinline__ v8f wmma_bf16(const FragBF& a, const FragBF& b, v8f c) {
  return __builtin_amdgcn_wmma_f32_16x16x32_bf16(false, a.v, false, b.v, (short)0, c, false, false);
}

// CDNA5 async global->LDS copy (16B per lane), tracked by ASYNCcnt.
// lds_off: wave-relative LDS byte address (low 32 bits of generic shared ptr).
__device__ __forceinline__ void async_copy16(unsigned lds_off, const void* gaddr) {
  asm volatile("global_load_async_to_lds_b128 %0, %1, off"
               :: "v"(lds_off), "v"(gaddr) : "memory");
}
__device__ __forceinline__ void async_wait0() {
  asm volatile("s_wait_asynccnt 0x0" ::: "memory");
}
template <typename T>
__device__ __forceinline__ unsigned lds_off_of(const T* p) {
  return (unsigned)(unsigned long long)p;
}

// ---------------- weight fp32 -> bf16 conversion ----------------
__global__ void k_wcvt(const float* __restrict__ Wf, const float* __restrict__ Wkv,
                       const float* __restrict__ Wo, unsigned short* __restrict__ Wfb,
                       unsigned short* __restrict__ Wkvb, unsigned short* __restrict__ Wob) {
  int i = blockIdx.x * 256 + threadIdx.x;
  if (i < CC * CC)        Wfb[i]  = f2bf(Wf[i]);
  if (i < 2 * INNER * CC) Wkvb[i] = f2bf(Wkv[i]);
  if (i < CC * INNER)     Wob[i]  = f2bf(Wo[i]);
}

// ---------------- command MLP + q ----------------
__global__ void k_mlp1(const float* __restrict__ ce, const float* __restrict__ W1,
                       const float* __restrict__ b1, float* __restrict__ hid) {
  int o = blockIdx.x * 256 + threadIdx.x;          // NB*512
  int b = o >> 9, j = o & 511;
  float s = b1[j];
  const float* cr = ce + (size_t)b * CMDD;
  const float* wr = W1 + (size_t)j * CMDD;
  for (int k = 0; k < CMDD; ++k) s += cr[k] * wr[k];
  hid[o] = fmaxf(s, 0.f);
}
__global__ void k_mlp2(const float* __restrict__ hid, const float* __restrict__ W2,
                       const float* __restrict__ b2, float* __restrict__ cmd) {
  int o = blockIdx.x * 256 + threadIdx.x;          // NB*CC
  int b = o >> 8, j = o & 255;
  float s = b2[j];
  const float* hr = hid + (size_t)b * 512;
  const float* wr = W2 + (size_t)j * 512;
  for (int k = 0; k < 512; ++k) s += hr[k] * wr[k];
  cmd[o] = s;
}
__global__ void k_mlp3(const float* __restrict__ cmd, const float* __restrict__ Wq,
                       float* __restrict__ q) {
  int o = blockIdx.x * 256 + threadIdx.x;          // NB*INNER
  int b = o >> 9, j = o & 511;
  float s = 0.f;
  const float* cr = cmd + (size_t)b * CC;
  const float* wr = Wq + (size_t)j * CC;
  for (int k = 0; k < CC; ++k) s += cr[k] * wr[k];
  q[o] = s * 0.125f;   // fold dim_head^-0.5
}

// qk[b,h,c] = sum_d q[b,h,d] * Wkv_k[h*64+d, c]   (k rows of Wkv = 0..511)
__global__ void k_qk(const float* __restrict__ q, const float* __restrict__ Wkv,
                     float* __restrict__ qk) {
  int i = blockIdx.x * 256 + threadIdx.x;          // NB*HEADS*CC
  int c = i & 255, bh = i >> 8, h = bh & 7, b = bh >> 3;
  float s = 0.f;
  for (int d = 0; d < DH; ++d)
    s += q[(size_t)b * INNER + h * DH + d] * Wkv[(size_t)(h * DH + d) * CC + c];
  qk[i] = s;
}

// ---------------- f = Wf*X + bf (WMMA) + dots = qk*f ----------------
__global__ __launch_bounds__(256) void k_fproj(
    const float* __restrict__ X, const unsigned short* __restrict__ Wfb,
    const float* __restrict__ bfv, const float* __restrict__ qk,
    float* __restrict__ fout, float* __restrict__ dots) {
  __shared__ unsigned short A_s[256][40];    // Wf tile, K-step 32, padded pitch
  __shared__ unsigned short B_s[64][40];     // features tile, pixel-major
  __shared__ unsigned short f_s[64][264];    // f tile bf16, pixel-major (for dots)
  const int t = threadIdx.x, lane = t & 31, w = t >> 5;
  const int half = lane >> 4, l15 = lane & 15;
  const int b = blockIdx.y, pix0 = blockIdx.x * 64;
  const int rowbase = (w >> 1) * 64, colbase = (w & 1) * 32;
  const v8f vz = {0.f,0.f,0.f,0.f,0.f,0.f,0.f,0.f};
  v8f acc[4][2];
  for (int i = 0; i < 4; ++i) for (int j = 0; j < 2; ++j) acc[i][j] = vz;

  for (int k0 = 0; k0 < CC; k0 += 32) {
    // async-copy Wf K-tile: row t = 64B = 4x16B chunks (no conversion needed)
    {
      const unsigned short* gp = Wfb + t * CC + k0;
      unsigned lo = lds_off_of(&A_s[t][0]);
      #pragma unroll
      for (int j = 0; j < 4; ++j) async_copy16(lo + j * 16, gp + j * 8);
    }
    // features tile (f32 -> bf16 in flight), pixel-major; prefetch next K-tile
    #pragma unroll
    for (int j = 0; j < 8; ++j) {
      int e = t + j * 256, kr = e >> 6, cx = e & 63;
      const float* gp = &X[((size_t)b * CC + k0 + kr) * HWPIX + pix0 + cx];
      B_s[cx][kr] = f2bf(*gp);
      if (cx == 0 && k0 + 32 < CC) __builtin_prefetch(gp + 32 * HWPIX, 0, 1);
    }
    async_wait0();
    __syncthreads();
    FragBF bfrag[2];
    #pragma unroll
    for (int tn = 0; tn < 2; ++tn) {
      int col = colbase + tn * 16 + l15;
      bfrag[tn].q[0] = *(const u32x4*)&B_s[col][half * 8];
      bfrag[tn].q[1] = *(const u32x4*)&B_s[col][16 + half * 8];
    }
    #pragma unroll
    for (int tm = 0; tm < 4; ++tm) {
      FragBF afrag;
      int row = rowbase + tm * 16 + l15;
      afrag.q[0] = *(const u32x4*)&A_s[row][half * 8];
      afrag.q[1] = *(const u32x4*)&A_s[row][16 + half * 8];
      #pragma unroll
      for (int tn = 0; tn < 2; ++tn)
        acc[tm][tn] = wmma_bf16(afrag, bfrag[tn], acc[tm][tn]);
    }
    __syncthreads();
  }
  // epilogue: bias, f32 -> global (d_out as f buffer), bf16 -> LDS for dots
  #pragma unroll
  for (int tm = 0; tm < 4; ++tm)
    #pragma unroll
    for (int tn = 0; tn < 2; ++tn) {
      int col = colbase + tn * 16 + l15;
      #pragma unroll
      for (int r = 0; r < 8; ++r) {
        int row = rowbase + tm * 16 + half * 8 + r;
        float v = acc[tm][tn][r] + bfv[row];
        fout[((size_t)b * CC + row) * HWPIX + pix0 + col] = v;
        f_s[col][row] = f2bf(v);
      }
    }
  __syncthreads();
  // dots[b,h,pix] = qk[b,h,:] . f[:,pix]
  #pragma unroll
  for (int j = 0; j < 2; ++j) {
    int o = t + j * 256, h = o >> 6, pix = o & 63;
    const float* qr = qk + ((size_t)b * HEADS + h) * CC;
    float s = 0.f;
    for (int c = 0; c < CC; ++c) s += qr[c] * bf2f(f_s[pix][c]);
    dots[((size_t)b * HEADS + h) * HWPIX + pix0 + pix] = s;
  }
}

// ---------------- softmax over HW per (b,h) ----------------
__global__ void k_softmax(const float* __restrict__ dots, float* __restrict__ attn) {
  __shared__ float red[256];
  int bh = blockIdx.x, t = threadIdx.x;
  const float* dr = dots + (size_t)bh * HWPIX;
  float* ar = attn + (size_t)bh * HWPIX;
  float m = -1e30f;
  for (int i = t; i < HWPIX; i += 256) m = fmaxf(m, dr[i]);
  red[t] = m; __syncthreads();
  for (int s = 128; s > 0; s >>= 1) { if (t < s) red[t] = fmaxf(red[t], red[t + s]); __syncthreads(); }
  m = red[0]; __syncthreads();
  float sum = 0.f;
  for (int i = t; i < HWPIX; i += 256) { float e = __expf(dr[i] - m); ar[i] = e; sum += e; }
  red[t] = sum; __syncthreads();
  for (int s = 128; s > 0; s >>= 1) { if (t < s) red[t] += red[t + s]; __syncthreads(); }
  float inv = 1.f / red[0];
  for (int i = t; i < HWPIX; i += 256) ar[i] *= inv;
}

// ---------------- v = Wkv_v*f ; scale by attn ; out = Wo*v + bo ; res = f + out ----------------
__global__ __launch_bounds__(256) void k_out(
    const unsigned short* __restrict__ Wkvb, const unsigned short* __restrict__ Wob,
    const float* __restrict__ bov, const float* __restrict__ attn,
    float* __restrict__ res) {
  __shared__ alignas(16) char reg1[20480];   // fs[32][264] (16896B) then A2[256][40] (20480B)
  __shared__ alignas(16) char reg2[40960];   // A1[512][40] (40960B) then V[32][520] (33280B)
  unsigned short (*fs)[264] = (unsigned short(*)[264])reg1;
  unsigned short (*A2)[40]  = (unsigned short(*)[40])reg1;
  unsigned short (*A1)[40]  = (unsigned short(*)[40])reg2;
  unsigned short (*V)[520]  = (unsigned short(*)[520])reg2;
  const int t = threadIdx.x, lane = t & 31, w = t >> 5;
  const int half = lane >> 4, l15 = lane & 15;
  const int b = blockIdx.y, pix0 = blockIdx.x * 32;
  const v8f vz = {0.f,0.f,0.f,0.f,0.f,0.f,0.f,0.f};

  // stage f tile bf16, pixel-major (res currently holds f)
  #pragma unroll
  for (int j = 0; j < 32; ++j) {
    int e = t + j * 256, row = e >> 5, col = e & 31;
    fs[col][row] = f2bf(res[((size_t)b * CC + row) * HWPIX + pix0 + col]);
  }
  __syncthreads();

  // GEMM1: v(512x32) = Wkv_v * f ; wave w owns head w (rows 64w..64w+63)
  v8f acc[4][2];
  for (int i = 0; i < 4; ++i) for (int j = 0; j < 2; ++j) acc[i][j] = vz;
  for (int k0 = 0; k0 < CC; k0 += 32) {
    // async-copy Wkv_v K-tile: 2 rows per thread, 4x16B chunks each
    #pragma unroll
    for (int rr = 0; rr < 2; ++rr) {
      int row = t * 2 + rr;
      const unsigned short* gp = Wkvb + (size_t)(INNER + row) * CC + k0;
      unsigned lo = lds_off_of(&A1[row][0]);
      #pragma unroll
      for (int j = 0; j < 4; ++j) async_copy16(lo + j * 16, gp + j * 8);
    }
    async_wait0();
    __syncthreads();
    FragBF bfrag[2];
    #pragma unroll
    for (int tn = 0; tn < 2; ++tn) {
      int col = tn * 16 + l15;
      bfrag[tn].q[0] = *(const u32x4*)&fs[col][k0 + half * 8];
      bfrag[tn].q[1] = *(const u32x4*)&fs[col][k0 + 16 + half * 8];
    }
    #pragma unroll
    for (int tm = 0; tm < 4; ++tm) {
      FragBF afrag;
      int row = w * 64 + tm * 16 + l15;
      afrag.q[0] = *(const u32x4*)&A1[row][half * 8];
      afrag.q[1] = *(const u32x4*)&A1[row][16 + half * 8];
      #pragma unroll
      for (int tn = 0; tn < 2; ++tn)
        acc[tm][tn] = wmma_bf16(afrag, bfrag[tn], acc[tm][tn]);
    }
    __syncthreads();
  }

  // scale rows by attn (head = wave id), store v bf16 pixel-major (A1 region now dead)
  const float* ar = attn + ((size_t)b * HEADS + w) * HWPIX + pix0;
  #pragma unroll
  for (int tn = 0; tn < 2; ++tn) {
    int col = tn * 16 + l15;
    float av = ar[col];
    #pragma unroll
    for (int tm = 0; tm < 4; ++tm)
      #pragma unroll
      for (int r = 0; r < 8; ++r) {
        int row = w * 64 + tm * 16 + half * 8 + r;
        V[col][row] = f2bf(acc[tm][tn][r] * av);
      }
  }
  __syncthreads();

  // GEMM2: out(256x32) = Wo * v ; wave w owns rows 32w..32w+31
  v8f oacc[2][2];
  for (int i = 0; i < 2; ++i) for (int j = 0; j < 2; ++j) oacc[i][j] = vz;
  for (int k0 = 0; k0 < INNER; k0 += 32) {
    // async-copy Wo K-tile: row t, 4x16B chunks
    {
      const unsigned short* gp = Wob + (size_t)t * INNER + k0;
      unsigned lo = lds_off_of(&A2[t][0]);
      #pragma unroll
      for (int j = 0; j < 4; ++j) async_copy16(lo + j * 16, gp + j * 8);
    }
    async_wait0();
    __syncthreads();
    FragBF bfrag[2];
    #pragma unroll
    for (int tn = 0; tn < 2; ++tn) {
      int col = tn * 16 + l15;
      bfrag[tn].q[0] = *(const u32x4*)&V[col][k0 + half * 8];
      bfrag[tn].q[1] = *(const u32x4*)&V[col][k0 + 16 + half * 8];
    }
    #pragma unroll
    for (int tm = 0; tm < 2; ++tm) {
      FragBF afrag;
      int row = w * 32 + tm * 16 + l15;
      afrag.q[0] = *(const u32x4*)&A2[row][half * 8];
      afrag.q[1] = *(const u32x4*)&A2[row][16 + half * 8];
      #pragma unroll
      for (int tn = 0; tn < 2; ++tn)
        oacc[tm][tn] = wmma_bf16(afrag, bfrag[tn], oacc[tm][tn]);
    }
    __syncthreads();
  }

  // epilogue: res = f + (out + bo), in-place on d_out
  #pragma unroll
  for (int tm = 0; tm < 2; ++tm)
    #pragma unroll
    for (int tn = 0; tn < 2; ++tn) {
      int col = tn * 16 + l15;
      #pragma unroll
      for (int r = 0; r < 8; ++r) {
        int c = w * 32 + tm * 16 + half * 8 + r;
        size_t g = ((size_t)b * CC + c) * HWPIX + pix0 + col;
        res[g] += oacc[tm][tn][r] + bov[c];
      }
    }
}

// ---------------- deterministic per-(b,c) stats ----------------
__global__ void k_stats(const float* __restrict__ res, float* __restrict__ gsum,
                        float* __restrict__ gsq) {
  __shared__ float s1[256], s2[256];
  int bc = blockIdx.x, t = threadIdx.x;
  const float* r = res + (size_t)bc * HWPIX;
  float a = 0.f, q = 0.f;
  for (int i = t; i < HWPIX; i += 256) { float v = r[i]; a += v; q += v * v; }
  s1[t] = a; s2[t] = q; __syncthreads();
  for (int s = 128; s > 0; s >>= 1) {
    if (t < s) { s1[t] += s1[t + s]; s2[t] += s2[t + s]; }
    __syncthreads();
  }
  if (t == 0) { gsum[bc] = s1[0]; gsq[bc] = s2[0]; }
}

__global__ void k_norm(float* __restrict__ out, const float* __restrict__ gsum,
                       const float* __restrict__ gsq) {
  size_t i = (size_t)blockIdx.x * 256 + threadIdx.x;
  if (i >= (size_t)NB * CC * HWPIX) return;
  int bc = (int)(i >> 12);
  float mean = gsum[bc] * (1.f / 4096.f);
  float var = gsq[bc] * (1.f / 4096.f) - mean * mean;
  out[i] = (out[i] - mean) * rsqrtf(var + 1e-5f);
}

extern "C" void kernel_launch(void* const* d_in, const int* in_sizes, int n_in,
                              void* d_out, int out_size, void* d_ws, size_t ws_size,
                              hipStream_t stream) {
  (void)in_sizes; (void)n_in; (void)out_size; (void)ws_size;
  const float* X   = (const float*)d_in[0];
  const float* ce  = (const float*)d_in[1];
  const float* Wf  = (const float*)d_in[2];
  const float* bfv = (const float*)d_in[3];
  const float* W1  = (const float*)d_in[4];
  const float* b1  = (const float*)d_in[5];
  const float* W2  = (const float*)d_in[6];
  const float* b2  = (const float*)d_in[7];
  const float* Wq  = (const float*)d_in[8];
  const float* Wkv = (const float*)d_in[9];
  const float* Wo  = (const float*)d_in[10];
  const float* bov = (const float*)d_in[11];
  float* out = (float*)d_out;
  char* ws = (char*)d_ws;

  float* q    = (float*)(ws + 0);            // 65536 B
  float* hid  = (float*)(ws + 65536);        // 65536 B
  float* cmdb = (float*)(ws + 131072);       // 32768 B
  float* qk   = (float*)(ws + 163840);       // 262144 B
  float* dots = (float*)(ws + 425984);       // 4 MiB
  float* attn = (float*)(ws + 4620288);      // 4 MiB
  float* gsum = (float*)(ws + 8814592);      // 32768 B
  float* gsq  = (float*)(ws + 8847360);      // 32768 B
  unsigned short* Wfb  = (unsigned short*)(ws + 8880128);   // 131072 B
  unsigned short* Wkvb = (unsigned short*)(ws + 9011200);   // 524288 B
  unsigned short* Wob  = (unsigned short*)(ws + 9535488);   // 262144 B

  k_wcvt<<<1024, 256, 0, stream>>>(Wf, Wkv, Wo, Wfb, Wkvb, Wob);
  k_mlp1<<<64, 256, 0, stream>>>(ce, W1, b1, hid);
  k_mlp2<<<32, 256, 0, stream>>>(hid, W2, b2, cmdb);
  k_mlp3<<<64, 256, 0, stream>>>(cmdb, Wq, q);
  k_qk<<<256, 256, 0, stream>>>(q, Wkv, qk);
  k_fproj<<<dim3(64, 32), 256, 0, stream>>>(X, Wfb, bfv, qk, out, dots);
  k_softmax<<<256, 256, 0, stream>>>(dots, attn);
  k_out<<<dim3(128, 32), 256, 0, stream>>>(Wkvb, Wob, bov, attn, out);
  k_stats<<<8192, 256, 0, stream>>>(out, gsum, gsq);
  k_norm<<<131072, 256, 0, stream>>>(out, gsum, gsq);
}
